// PillarContext3D_dsa_53498112639176
// MI455X (gfx1250) — matrix operand order repaired and checked by reading, not compiled, not run on hardware
//
#include <hip/hip_runtime.h>
#include <hip/hip_bf16.h>

// ---------------------------------------------------------------------------
// Problem constants (from reference)
// ---------------------------------------------------------------------------
#define BB 2
#define NP 16000
#define MKP 2048
#define CC 64
#define NXg 432
#define NYg 496
#define GG 4
#define NTOK 2048          // n for attention
#define NS 16              // neighbors per query

typedef __attribute__((ext_vector_type(16))) _Float16 v16h;
typedef __attribute__((ext_vector_type(8)))  _Float16 v8h;
typedef __attribute__((ext_vector_type(4)))  _Float16 v4h;
typedef __attribute__((ext_vector_type(8)))  float    v8f;
typedef __attribute__((ext_vector_type(4)))  float    v4f;

// ---------------------------------------------------------------------------
// Generic WMMA GEMM:  C(MxN) = act( opA(A) * opB(B) + bias )
//   opA(A)[m,k] = TA ? A[k*lda+m] : A[m*lda+k]
//   opB(B)[k,n] = TB ? B[n*ldb+k] : B[k*ldb+n]
//
// f32 inputs are converted to f16 while staging into LDS; staging writes are
// PERMUTED into v_wmma fragment order so each lane's 16 fragment elements are
// contiguous (two aligned ds_load_b128 per fragment):
//   A (16x32) lane(m,h): element e -> k with h=(k>>3)&1, e=(k&7)+((k>>4)&1)*8
//   B (32x16) lane(n,h): element e -> k with h=(k>>3)&1, e=(k&7)+((k>>4)&1)*8
// TA/TB are template params so each variant's contiguous global axis is
// loaded with float4 (global_load_b128); 4-aligned k-chunks stay contiguous
// under the permutation, enabling 8-byte v4h LDS stores on the fast paths.
//
// Block = 128 threads = 4 waves, 64x64 C tile, waves 2x2; each wave owns a
// 32x32 quadrant = 2x2 WMMA tiles -> 4 v_wmma per 32-wide K step with 2x
// fragment reuse. M,N multiples of 64; K multiple of 32 (true everywhere).
// ---------------------------------------------------------------------------
template <int TA, int TB>
__global__ __launch_bounds__(128) void k_wmma_gemm(
    const float* __restrict__ A, const float* __restrict__ B,
    float* __restrict__ C, int M, int N, int K,
    int lda, int ldb, int ldc,
    const float* __restrict__ bias, int relu)
{
  // rows padded to 40 halfs (80B, 16B-aligned) to spread LDS banks
  __shared__ _Float16 As[64][40];   // As[r][h*16+e]  (r = tile row, permuted k)
  __shared__ _Float16 Bs[64][40];   // Bs[n][h*16+e]  (n = tile col, permuted k)

  const int tid  = threadIdx.x;
  const int lane = tid & 31;
  const int w    = tid >> 5;            // wave id 0..3
  const int bm   = blockIdx.y * 64;
  const int bn   = blockIdx.x * 64;
  const int wm   = (w >> 1) * 32;       // wave's 32x32 quadrant
  const int wn   = (w & 1) * 32;

  const int mloc = lane & 15;
  const int half = lane >> 4;           // 0 or 1

  v8f acc[2][2] = {};

  for (int k0 = 0; k0 < K; k0 += 32) {
    // ---- stage A tile 64x32 (fragment-permuted) ----
    if (TA == 0) {
      // contiguous along k: float4 load + single v4h store
      for (int e = tid; e < 64 * 8; e += 128) {
        int r  = e >> 3;
        int c4 = (e & 7) * 4;
        v4f va = *(const v4f*)(A + (size_t)(bm + r) * lda + (k0 + c4));
        int h  = (c4 >> 3) & 1;
        int eo = (c4 & 7) + ((c4 >> 4) & 1) * 8;
        v4h hv = { (_Float16)va.x, (_Float16)va.y, (_Float16)va.z, (_Float16)va.w };
        *(v4h*)&As[r][h * 16 + eo] = hv;
      }
    } else {
      // A[k*lda+m]: contiguous along m: float4 load + 4 b16 scatters
      for (int e = tid; e < 16 * 32; e += 128) {
        int k  = e & 31;
        int r4 = (e >> 5) * 4;
        v4f va = *(const v4f*)(A + (size_t)(k0 + k) * lda + (bm + r4));
        int h  = (k >> 3) & 1;
        int eo = (k & 7) + ((k >> 4) & 1) * 8;
        As[r4 + 0][h * 16 + eo] = (_Float16)va.x;
        As[r4 + 1][h * 16 + eo] = (_Float16)va.y;
        As[r4 + 2][h * 16 + eo] = (_Float16)va.z;
        As[r4 + 3][h * 16 + eo] = (_Float16)va.w;
      }
    }
    // ---- stage B tile 32x64 (fragment-permuted) ----
    if (TB == 0) {
      // B[k*ldb+n]: contiguous along n: float4 load + 4 b16 scatters
      for (int e = tid; e < 32 * 16; e += 128) {
        int kr = e >> 4;
        int n4 = (e & 15) * 4;
        v4f vb = *(const v4f*)(B + (size_t)(k0 + kr) * ldb + (bn + n4));
        int h  = (kr >> 3) & 1;
        int eo = (kr & 7) + ((kr >> 4) & 1) * 8;
        Bs[n4 + 0][h * 16 + eo] = (_Float16)vb.x;
        Bs[n4 + 1][h * 16 + eo] = (_Float16)vb.y;
        Bs[n4 + 2][h * 16 + eo] = (_Float16)vb.z;
        Bs[n4 + 3][h * 16 + eo] = (_Float16)vb.w;
      }
    } else {
      // B[n*ldb+k]: contiguous along k: float4 load + single v4h store
      for (int e = tid; e < 64 * 8; e += 128) {
        int n  = e >> 3;
        int k4 = (e & 7) * 4;
        v4f vb = *(const v4f*)(B + (size_t)(bn + n) * ldb + (k0 + k4));
        int h  = (k4 >> 3) & 1;
        int eo = (k4 & 7) + ((k4 >> 4) & 1) * 8;
        v4h hv = { (_Float16)vb.x, (_Float16)vb.y, (_Float16)vb.z, (_Float16)vb.w };
        *(v4h*)&Bs[n][h * 16 + eo] = hv;
      }
    }
    __syncthreads();

    // ---- fragment loads: contiguous 32B per fragment (2x ds_load_b128) ----
    v16h afr[2], bfr[2];
#pragma unroll
    for (int i = 0; i < 2; ++i) {
      const _Float16* ap = &As[wm + i * 16 + mloc][half * 16];
      v8h a0 = *(const v8h*)(ap);
      v8h a1 = *(const v8h*)(ap + 8);
      afr[i] = __builtin_shufflevector(a0, a1, 0, 1, 2, 3, 4, 5, 6, 7,
                                       8, 9, 10, 11, 12, 13, 14, 15);
      const _Float16* bp = &Bs[wn + i * 16 + mloc][half * 16];
      v8h b0 = *(const v8h*)(bp);
      v8h b1 = *(const v8h*)(bp + 8);
      bfr[i] = __builtin_shufflevector(b0, b1, 0, 1, 2, 3, 4, 5, 6, 7,
                                       8, 9, 10, 11, 12, 13, 14, 15);
    }

    // ---- 4 WMMAs with 2x fragment reuse ----
#pragma unroll
    for (int i = 0; i < 2; ++i)
#pragma unroll
      for (int j = 0; j < 2; ++j)
        acc[i][j] = __builtin_amdgcn_wmma_f32_16x16x32_f16(
            false, afr[i], false, bfr[j], (short)0, acc[i][j], false, false);

    __syncthreads();
  }

  // ---- store: C/D layout lane n=lane&15; VGPR v -> row v + 8*(lane>>4) ----
#pragma unroll
  for (int i = 0; i < 2; ++i) {
#pragma unroll
    for (int j = 0; j < 2; ++j) {
#pragma unroll
      for (int v = 0; v < 8; ++v) {
        int m = bm + wm + i * 16 + v + 8 * half;
        int n = bn + wn + j * 16 + mloc;
        float val = acc[i][j][v];
        if (bias) val += bias[m];
        if (relu) val = fmaxf(val, 0.f);
        C[(size_t)m * ldc + n] = val;
      }
    }
  }
}

// ---------------------------------------------------------------------------
// Pillar centers from voxel coords
// ---------------------------------------------------------------------------
__global__ void k_centers(const int* __restrict__ coords, float* __restrict__ centers)
{
  int i = blockIdx.x * 256 + threadIdx.x;
  if (i >= BB * NP) return;
  float z = (float)coords[i * 4 + 1];
  float y = (float)coords[i * 4 + 2];
  float x = (float)coords[i * 4 + 3];
  centers[i * 3 + 0] = x * 0.16f + 0.08f;            // VX/2 + X0
  centers[i * 3 + 1] = y * 0.16f + (0.08f - 39.68f); // VY/2 + Y0
  centers[i * 3 + 2] = z * 4.0f + (2.0f - 3.0f);     // VZ/2 + Z0
}

// ---------------------------------------------------------------------------
// Furthest point sampling: one workgroup (512 threads) per batch.
// Sequential 2048-iteration loop with LDS argmax tree (first-index ties).
// ---------------------------------------------------------------------------
#define FPS_T 512
__global__ __launch_bounds__(FPS_T) void k_fps(
    const float* __restrict__ centers, float* __restrict__ dist, int* __restrict__ kpidx)
{
  int b = blockIdx.x;
  const float* cx = centers + (size_t)b * NP * 3;
  float* db = dist + (size_t)b * NP;
  int* ib = kpidx + (size_t)b * MKP;

  __shared__ float sval[FPS_T];
  __shared__ int   sidx[FPS_T];
  __shared__ int   s_far;
  __shared__ float fx, fy, fz;

  int tid = threadIdx.x;
  for (int j = tid; j < NP; j += FPS_T) db[j] = 1e10f;
  if (tid == 0) s_far = 0;
  __syncthreads();

  for (int i = 0; i < MKP; ++i) {
    if (tid == 0) {
      int far = s_far;
      ib[i] = far;
      fx = cx[far * 3 + 0]; fy = cx[far * 3 + 1]; fz = cx[far * 3 + 2];
    }
    __syncthreads();
    float px = fx, py = fy, pz = fz;
    float best = -1.f; int bidx = 0x7fffffff;
    for (int j = tid; j < NP; j += FPS_T) {
      float dx = cx[j * 3 + 0] - px;
      float dy = cx[j * 3 + 1] - py;
      float dz = cx[j * 3 + 2] - pz;
      float d = dx * dx + dy * dy + dz * dz;
      float nd = db[j];
      nd = d < nd ? d : nd;
      db[j] = nd;
      if (nd > best) { best = nd; bidx = j; }
    }
    sval[tid] = best; sidx[tid] = bidx;
    __syncthreads();
    for (int s = FPS_T / 2; s > 0; s >>= 1) {
      if (tid < s) {
        float ov = sval[tid + s]; int oi = sidx[tid + s];
        if (ov > sval[tid] || (ov == sval[tid] && oi < sidx[tid])) {
          sval[tid] = ov; sidx[tid] = oi;
        }
      }
      __syncthreads();
    }
    if (tid == 0) s_far = sidx[0];
    __syncthreads();
  }
}

__global__ void k_gather_kp(const float* __restrict__ centers,
                            const int* __restrict__ kpidx, float* __restrict__ keyp)
{
  int i = blockIdx.x * 256 + threadIdx.x;
  if (i >= BB * MKP) return;
  int b = i / MKP;
  int s = kpidx[i];
  size_t src = ((size_t)b * NP + s) * 3;
  keyp[i * 3 + 0] = centers[src + 0];
  keyp[i * 3 + 1] = centers[src + 1];
  keyp[i * 3 + 2] = centers[src + 2];
}

// ---------------------------------------------------------------------------
// Radius grouping + shared MLP (67->64->64, ReLU) + masked maxpool.
// One 64-thread (2-wave) block per query. Per-thread insertion top-16 over a
// strided slice, single-thread LDS merge (first-index ties == lax.top_k),
// then cooperative MLP: thread t owns hidden/output channel t.
// featChMajor: 0 -> sfeat[(b*ns+s)*64+f]; 1 -> sfeat[(b*64+f)*ld + s]
// ---------------------------------------------------------------------------
__global__ __launch_bounds__(64) void k_group_pool(
    const float* __restrict__ qxyz, int nq, const float* __restrict__ sxyz, int ns,
    const float* __restrict__ sfeat, int featChMajor, int featLd, float r2,
    const float* __restrict__ W1, const float* __restrict__ b1,
    const float* __restrict__ W2, const float* __restrict__ b2,
    float* __restrict__ outp)
{
  int b = blockIdx.x / nq;
  int q = blockIdx.x % nq;
  const float* src = sxyz + (size_t)b * ns * 3;
  const float* qp  = qxyz + ((size_t)b * nq + q) * 3;
  float qx = qp[0], qy = qp[1], qz = qp[2];
  int tid = threadIdx.x;

  float td[NS]; int ti[NS];
#pragma unroll
  for (int r = 0; r < NS; ++r) { td[r] = 3e38f; ti[r] = 0x7fffffff; }

  for (int j = tid; j < ns; j += 64) {
    float dx = src[j * 3 + 0] - qx;
    float dy = src[j * 3 + 1] - qy;
    float dz = src[j * 3 + 2] - qz;
    float d = dx * dx + dy * dy + dz * dz;
    if (d < td[NS - 1]) {
      int r = NS - 1;
      while (r > 0 && td[r - 1] > d) { td[r] = td[r - 1]; ti[r] = ti[r - 1]; --r; }
      td[r] = d; ti[r] = j;
    }
  }

  __shared__ float md[64 * NS];
  __shared__ int   mi[64 * NS];
  __shared__ float bd[NS];
  __shared__ int   bi[NS];
  __shared__ int   vld[NS];
  __shared__ float feat[NS][68];
  __shared__ float h1[NS][64];

#pragma unroll
  for (int r = 0; r < NS; ++r) { md[tid * NS + r] = td[r]; mi[tid * NS + r] = ti[r]; }
  __syncthreads();

  if (tid == 0) {
    for (int r = 0; r < NS; ++r) {
      float bv = 3e38f; int bj = 0x7fffffff, bpos = 0;
      for (int e = 0; e < 64 * NS; ++e) {
        float v = md[e]; int vi = mi[e];
        if (v < bv || (v == bv && vi < bj)) { bv = v; bj = vi; bpos = e; }
      }
      bd[r] = bv; bi[r] = bj;
      md[bpos] = 3e38f; mi[bpos] = 0x7fffffff;
    }
  }
  __syncthreads();

  for (int e = tid; e < NS * 67; e += 64) {
    int nb = e / 67, f = e % 67;
    int s = bi[nb];
    float v;
    if (f < 3) {
      v = src[s * 3 + f] - (f == 0 ? qx : (f == 1 ? qy : qz));
    } else if (featChMajor) {
      v = sfeat[((size_t)b * 64 + (f - 3)) * featLd + s];
    } else {
      v = sfeat[((size_t)b * ns + s) * 64 + (f - 3)];
    }
    feat[nb][f] = v;
  }
  if (tid < NS) vld[tid] = (bd[tid] <= r2) || (tid == 0);
  __syncthreads();

  float bia = b1[tid];
  for (int nb = 0; nb < NS; ++nb) {
    float acc = bia;
    for (int i = 0; i < 67; ++i) acc += feat[nb][i] * W1[i * 64 + tid];
    h1[nb][tid] = fmaxf(acc, 0.f);
  }
  __syncthreads();

  float bib = b2[tid];
  float best = -1e9f;
  for (int nb = 0; nb < NS; ++nb) {
    float acc = bib;
    for (int i = 0; i < 64; ++i) acc += h1[nb][i] * W2[i * 64 + tid];
    acc = fmaxf(acc, 0.f);
    best = fmaxf(best, vld[nb] ? acc : -1e9f);
  }
  outp[((size_t)b * nq + q) * 64 + tid] = best;
}

// ---------------------------------------------------------------------------
// local features (M,64) -> x0 (64, 2048) channel-major
// ---------------------------------------------------------------------------
__global__ void k_transpose_local(const float* __restrict__ lf, float* __restrict__ x0)
{
  int e = blockIdx.x * 256 + threadIdx.x;
  if (e >= MKP * 64) return;
  int m = e >> 6, ch = e & 63;
  x0[(size_t)ch * NTOK + m] = lf[e];
}

// ---------------------------------------------------------------------------
// row softmax over 2048-wide rows
// ---------------------------------------------------------------------------
__global__ __launch_bounds__(256) void k_softmax(float* __restrict__ S)
{
  int row = blockIdx.x;
  float* r = S + (size_t)row * NTOK;
  __shared__ float red[256];
  int tid = threadIdx.x;
  float mx = -3e38f;
  for (int j = tid; j < NTOK; j += 256) mx = fmaxf(mx, r[j]);
  red[tid] = mx; __syncthreads();
  for (int s = 128; s > 0; s >>= 1) { if (tid < s) red[tid] = fmaxf(red[tid], red[tid + s]); __syncthreads(); }
  mx = red[0]; __syncthreads();
  float sum = 0.f;
  for (int j = tid; j < NTOK; j += 256) { float e = expf(r[j] - mx); r[j] = e; sum += e; }
  red[tid] = sum; __syncthreads();
  for (int s = 128; s > 0; s >>= 1) { if (tid < s) red[tid] += red[tid + s]; __syncthreads(); }
  float inv = 1.f / red[0];
  for (int j = tid; j < NTOK; j += 256) r[j] *= inv;
}

// ---------------------------------------------------------------------------
// grouped 1x1 conv: z[g,o,:] = sum_i Wz[g,o,i] * in[g,i,:]
// ---------------------------------------------------------------------------
__global__ void k_gconv(const float* __restrict__ Wz, const float* __restrict__ inp,
                        float* __restrict__ zb, int cg)
{
  int j = blockIdx.y * 256 + threadIdx.x;
  int co = blockIdx.x;
  int g = co / cg, o = co % cg;
  const float* w  = Wz + ((size_t)g * cg + o) * cg;
  const float* ib = inp + (size_t)g * cg * NTOK;
  float acc = 0.f;
  for (int i = 0; i < cg; ++i) acc += w[i] * ib[(size_t)i * NTOK + j];
  zb[(size_t)co * NTOK + j] = acc;
}

__global__ __launch_bounds__(256) void k_gnstats(const float* __restrict__ zb,
                                                 float* __restrict__ st, int cg)
{
  int g = blockIdx.x, tid = threadIdx.x;
  const float* base = zb + (size_t)g * cg * NTOK;
  int cnt = cg * NTOK;
  float s = 0.f, s2 = 0.f;
  for (int e = tid; e < cnt; e += 256) { float v = base[e]; s += v; s2 += v * v; }
  __shared__ float rs[256], rq[256];
  rs[tid] = s; rq[tid] = s2; __syncthreads();
  for (int sh = 128; sh > 0; sh >>= 1) {
    if (tid < sh) { rs[tid] += rs[tid + sh]; rq[tid] += rq[tid + sh]; }
    __syncthreads();
  }
  if (tid == 0) {
    float mu = rs[0] / (float)cnt;
    float var = rq[0] / (float)cnt - mu * mu;
    st[g * 2 + 0] = mu;
    st[g * 2 + 1] = rsqrtf(var + 1e-5f);
  }
}

__global__ void k_gnapply(const float* __restrict__ zb, const float* __restrict__ st,
                          const float* __restrict__ gamma, const float* __restrict__ beta,
                          const float* __restrict__ xin, float* __restrict__ dst,
                          int c, int cg)
{
  int idx = blockIdx.x * 256 + threadIdx.x;
  if (idx >= c * NTOK) return;
  int ci = idx >> 11;            // / NTOK
  int g = ci / cg;
  float mu = st[g * 2], rstd = st[g * 2 + 1];
  dst[idx] = (zb[idx] - mu) * rstd * gamma[ci] + beta[ci] + xin[idx];
}

// ---------------------------------------------------------------------------
// deterministic BEV scatter: winner map via atomicMax (last pillar index wins)
// ---------------------------------------------------------------------------
__global__ void k_scatter_win(const int* __restrict__ coords, int* __restrict__ wmap)
{
  int i = blockIdx.x * 256 + threadIdx.x;
  if (i >= BB * NP) return;
  int bb = coords[i * 4 + 0];
  int y = coords[i * 4 + 2], x = coords[i * 4 + 3];
  atomicMax(&wmap[((size_t)bb * NYg + y) * NXg + x], i + 1);
}

__global__ void k_scatter(const int* __restrict__ coords, const int* __restrict__ wmap,
                          const float* __restrict__ dec, float* __restrict__ cpf)
{
  int i = blockIdx.x * 256 + threadIdx.x;
  if (i >= BB * NP) return;
  int bb = coords[i * 4 + 0];
  int y = coords[i * 4 + 2], x = coords[i * 4 + 3];
  if (wmap[((size_t)bb * NYg + y) * NXg + x] != i + 1) return;
  for (int ch = 0; ch < 64; ++ch)
    cpf[(((size_t)bb * 64 + ch) * NYg + y) * NXg + x] = dec[(size_t)i * 64 + ch];
}

// ---------------------------------------------------------------------------
// antialiased bilinear (triangle) downsample by integer factor f,
// per-axis weight renormalization at borders (jax.image.resize semantics)
// ---------------------------------------------------------------------------
__global__ void k_resize(const float* __restrict__ in, float* __restrict__ outp,
                         int f, int OH, int OW)
{
  int idx = blockIdx.x * 256 + threadIdx.x;
  int total = BB * 64 * OH * OW;
  if (idx >= total) return;
  int ox = idx % OW; int t = idx / OW;
  int oy = t % OH;  int pc = t / OH;       // b*64+ch plane
  const float* plane = in + (size_t)pc * NYg * NXg;
  float py = (oy + 0.5f) * f - 0.5f;
  float px = (ox + 0.5f) * f - 0.5f;
  int y0 = (int)floorf(py) - f + 1, y1 = (int)floorf(py) + f;
  int x0 = (int)floorf(px) - f + 1, x1 = (int)floorf(px) + f;
  float wys = 0.f, wxs = 0.f, acc = 0.f;
  for (int iy = y0; iy <= y1; ++iy) {
    if (iy < 0 || iy >= NYg) continue;
    float wy = 1.f - fabsf((float)iy - py) / (float)f;
    if (wy > 0.f) wys += wy;
  }
  for (int ix = x0; ix <= x1; ++ix) {
    if (ix < 0 || ix >= NXg) continue;
    float wx = 1.f - fabsf((float)ix - px) / (float)f;
    if (wx > 0.f) wxs += wx;
  }
  for (int iy = y0; iy <= y1; ++iy) {
    if (iy < 0 || iy >= NYg) continue;
    float wy = 1.f - fabsf((float)iy - py) / (float)f;
    if (wy <= 0.f) continue;
    for (int ix = x0; ix <= x1; ++ix) {
      if (ix < 0 || ix >= NXg) continue;
      float wx = 1.f - fabsf((float)ix - px) / (float)f;
      if (wx <= 0.f) continue;
      acc += wy * wx * plane[(size_t)iy * NXg + ix];
    }
  }
  outp[idx] = acc / (wys * wxs);
}

// ---------------------------------------------------------------------------
// Host-side helpers
// ---------------------------------------------------------------------------
static inline void launch_gemm(hipStream_t s, const float* A, const float* B, float* C,
                               int M, int N, int K, int lda, int ldb, int ldc,
                               int tA, int tB, const float* bias, int relu)
{
  dim3 grid(N / 64, M / 64);
  if (!tA && !tB)
    k_wmma_gemm<0, 0><<<grid, 128, 0, s>>>(A, B, C, M, N, K, lda, ldb, ldc, bias, relu);
  else if (tA && !tB)
    k_wmma_gemm<1, 0><<<grid, 128, 0, s>>>(A, B, C, M, N, K, lda, ldb, ldc, bias, relu);
  else
    k_wmma_gemm<0, 1><<<grid, 128, 0, s>>>(A, B, C, M, N, K, lda, ldb, ldc, bias, relu);
}

struct SaW { const float *Wt, *Wp, *Wg, *Wz, *g, *b; int c; };

static void run_sa(hipStream_t s, const SaW& sa, const float* xin, float* dst,
                   float* tb, float* pb, float* gb, float* ob, float* zb,
                   float* attn, float* gnst)
{
  int c = sa.c, cg = c / GG;
  // q/k/v projections
  launch_gemm(s, sa.Wt, xin, tb, c, NTOK, c, c, NTOK, NTOK, 0, 0, nullptr, 0);
  launch_gemm(s, sa.Wp, xin, pb, c, NTOK, c, c, NTOK, NTOK, 0, 0, nullptr, 0);
  launch_gemm(s, sa.Wg, xin, gb, c, NTOK, c, c, NTOK, NTOK, 0, 0, nullptr, 0);
  // attn logits = t^T @ p  (n x n)
  launch_gemm(s, tb, pb, attn, NTOK, NTOK, c, NTOK, NTOK, NTOK, 1, 0, nullptr, 0);
  k_softmax<<<NTOK, 256, 0, s>>>(attn);
  // out[c,m] = sum_n g[c,n]*attn[m,n]  ->  g @ attn^T
  launch_gemm(s, gb, attn, ob, c, NTOK, NTOK, NTOK, NTOK, NTOK, 0, 1, nullptr, 0);
  // grouped conv z
  k_gconv<<<dim3(c, NTOK / 256), 256, 0, s>>>(sa.Wz, ob, zb, cg);
  // GroupNorm + affine + residual
  k_gnstats<<<GG, 256, 0, s>>>(zb, gnst, cg);
  k_gnapply<<<(c * NTOK + 255) / 256, 256, 0, s>>>(zb, gnst, sa.g, sa.b, xin, dst, c, cg);
}

extern "C" void kernel_launch(void* const* d_in, const int* in_sizes, int n_in,
                              void* d_out, int out_size, void* d_ws, size_t ws_size,
                              hipStream_t stream)
{
  (void)in_sizes; (void)n_in; (void)out_size; (void)ws_size;

  const float* pf     = (const float*)d_in[0];
  const int*   coords = (const int*)d_in[1];
  const float* W_loc1 = (const float*)d_in[2];
  const float* b_loc1 = (const float*)d_in[3];
  const float* W_loc2 = (const float*)d_in[4];
  const float* b_loc2 = (const float*)d_in[5];
  SaW sa[3];
  for (int s = 0; s < 3; ++s) {
    sa[s].Wt = (const float*)d_in[6 + s * 6 + 0];
    sa[s].Wp = (const float*)d_in[6 + s * 6 + 1];
    sa[s].Wg = (const float*)d_in[6 + s * 6 + 2];
    sa[s].Wz = (const float*)d_in[6 + s * 6 + 3];
    sa[s].g  = (const float*)d_in[6 + s * 6 + 4];
    sa[s].b  = (const float*)d_in[6 + s * 6 + 5];
  }
  sa[0].c = CC; sa[1].c = 2 * CC; sa[2].c = 2 * CC;
  const float* W_red1 = (const float*)d_in[24];
  const float* b_red1 = (const float*)d_in[25];
  const float* W_red2 = (const float*)d_in[26];
  const float* b_red2 = (const float*)d_in[27];
  const float* W_dec1 = (const float*)d_in[28];
  const float* b_dec1 = (const float*)d_in[29];
  const float* W_dec2 = (const float*)d_in[30];
  const float* b_dec2 = (const float*)d_in[31];

  float* out = (float*)d_out;

  // ---------------- workspace layout (~148 MB) ----------------
  char* ws = (char*)d_ws;
  size_t off = 0;
  auto alloc = [&](size_t bytes) -> void* {
    void* p = ws + off;
    off += (bytes + 255) & ~(size_t)255;
    return p;
  };
  float* centers = (float*)alloc((size_t)BB * NP * 3 * 4);
  float* dist    = (float*)alloc((size_t)BB * NP * 4);
  int*   kpidx   = (int*)  alloc((size_t)BB * MKP * 4);
  float* keyp    = (float*)alloc((size_t)BB * MKP * 3 * 4);
  float* localf  = (float*)alloc((size_t)BB * MKP * CC * 4);
  float* xbuf    = (float*)alloc((size_t)2 * CC * NTOK * 4);      // [x0 ; gf/a1r]
  float* abuf    = (float*)alloc((size_t)2 * CC * NTOK * 4);      // sa1/sa2 output
  float* tb      = (float*)alloc((size_t)2 * CC * NTOK * 4);
  float* pb      = (float*)alloc((size_t)2 * CC * NTOK * 4);
  float* gb      = (float*)alloc((size_t)2 * CC * NTOK * 4);
  float* ob      = (float*)alloc((size_t)2 * CC * NTOK * 4);
  float* zb      = (float*)alloc((size_t)2 * CC * NTOK * 4);
  float* attn    = (float*)alloc((size_t)NTOK * NTOK * 4);
  float* hred    = (float*)alloc((size_t)CC * NTOK * 4);
  float* gnst    = (float*)alloc((size_t)GG * 2 * 4);
  float* ctx     = (float*)alloc((size_t)BB * CC * NTOK * 4);     // per-batch (64,2048)
  float* dec     = (float*)alloc((size_t)BB * NP * CC * 4);
  int*   wmap    = (int*)  alloc((size_t)BB * NYg * NXg * 4);
  float* cpf     = (float*)alloc((size_t)BB * CC * NYg * NXg * 4);

  // ---------------- pipeline ----------------
  k_centers<<<(BB * NP + 255) / 256, 256, 0, stream>>>(coords, centers);
  k_fps<<<BB, FPS_T, 0, stream>>>(centers, dist, kpidx);
  k_gather_kp<<<(BB * MKP + 255) / 256, 256, 0, stream>>>(centers, kpidx, keyp);

  // local grouping: 2048 queries vs 16000 sources, r=2
  k_group_pool<<<BB * MKP, 64, 0, stream>>>(
      keyp, MKP, centers, NP, pf, /*chMajor=*/0, /*ld=*/0, 4.0f,
      W_loc1, b_loc1, W_loc2, b_loc2, localf);

  for (int b = 0; b < BB; ++b) {
    k_transpose_local<<<(MKP * CC + 255) / 256, 256, 0, stream>>>(
        localf + (size_t)b * MKP * CC, xbuf);
    // sa0 on x0 (c=64); write gf into concat rows 64..127
    run_sa(stream, sa[0], xbuf, xbuf + (size_t)CC * NTOK, tb, pb, gb, ob, zb, attn, gnst);
    // sa1 on [x0; gf] (c=128)
    run_sa(stream, sa[1], xbuf, abuf, tb, pb, gb, ob, zb, attn, gnst);
    // reduce_dim(a1) -> rows 64..127 of xbuf
    launch_gemm(stream, W_red1, abuf, hred, CC, NTOK, 2 * CC, 2 * CC, NTOK, NTOK, 0, 0, b_red1, 1);
    launch_gemm(stream, W_red2, hred, xbuf + (size_t)CC * NTOK, CC, NTOK, CC, CC, NTOK, NTOK, 0, 0, b_red2, 1);
    // sa2 on [x0; a1r]
    run_sa(stream, sa[2], xbuf, abuf, tb, pb, gb, ob, zb, attn, gnst);
    // reduce_dim(a2) -> ctx[b] (64, 2048)
    launch_gemm(stream, W_red1, abuf, hred, CC, NTOK, 2 * CC, 2 * CC, NTOK, NTOK, 0, 0, b_red1, 1);
    launch_gemm(stream, W_red2, hred, ctx + (size_t)b * CC * NTOK, CC, NTOK, CC, CC, NTOK, NTOK, 0, 0, b_red2, 1);
  }

  // decode grouping: 16000 queries vs 2048 keypoints, r=4, features ch-major
  k_group_pool<<<BB * NP, 64, 0, stream>>>(
      centers, NP, keyp, MKP, ctx, /*chMajor=*/1, /*ld=*/NTOK, 16.0f,
      W_dec1, b_dec1, W_dec2, b_dec2, dec);

  // BEV scatter (deterministic winner) + pyramid
  hipMemsetAsync(wmap, 0, (size_t)BB * NYg * NXg * 4, stream);
  hipMemsetAsync(cpf, 0, (size_t)BB * CC * NYg * NXg * 4, stream);
  k_scatter_win<<<(BB * NP + 255) / 256, 256, 0, stream>>>(coords, wmap);
  k_scatter<<<(BB * NP + 255) / 256, 256, 0, stream>>>(coords, wmap, dec, cpf);

  const size_t P1 = (size_t)BB * CC * (NYg / 2) * (NXg / 2);
  const size_t P2 = (size_t)BB * CC * (NYg / 4) * (NXg / 4);
  const size_t P3 = (size_t)BB * CC * (NYg / 8) * (NXg / 8);
  k_resize<<<(int)((P1 + 255) / 256), 256, 0, stream>>>(cpf, out, 2, NYg / 2, NXg / 2);
  k_resize<<<(int)((P2 + 255) / 256), 256, 0, stream>>>(cpf, out + P1, 4, NYg / 4, NXg / 4);
  k_resize<<<(int)((P3 + 255) / 256), 256, 0, stream>>>(cpf, out + P1 + P2, 8, NYg / 8, NXg / 8);
}
